// ReasonerModule_3118146257673
// MI455X (gfx1250) — matrix operand order repaired and checked by reading, not compiled, not run on hardware
//
#include <hip/hip_runtime.h>

#define N_ELEM 128
#define N_DESC 2048
#define KPD    4
#define BATCHN 8192
#define NCOL   (N_DESC * KPD)   // 8192 flattened coef columns

typedef __bf16 bf16_t;
typedef bf16_t v16bf __attribute__((ext_vector_type(16)));
typedef bf16_t v8bf  __attribute__((ext_vector_type(8)));
typedef float  v8f   __attribute__((ext_vector_type(8)));

__device__ __forceinline__ unsigned short f2bf_rne(float f) {
    unsigned int u = __float_as_uint(f);
    u += 0x7FFFu + ((u >> 16) & 1u);   // round-to-nearest-even
    return (unsigned short)(u >> 16);
}

// ---------------- Kernel 1: L2-normalize rows of x, emit bf16 -------------
// one wave (32 lanes) per row of 128 floats; 8 waves / block
__global__ void __launch_bounds__(256) k_normalize(const float* __restrict__ x,
                                                   unsigned short* __restrict__ xn) {
    const int wave = threadIdx.x >> 5;
    const int lane = threadIdx.x & 31;
    const int row  = blockIdx.x * 8 + wave;
    float4 v = ((const float4*)(x + row * N_ELEM))[lane];
    float ss = v.x * v.x + v.y * v.y + v.z * v.z + v.w * v.w;
#pragma unroll
    for (int off = 16; off > 0; off >>= 1) ss += __shfl_xor(ss, off, 32);
    const float scale = 1.0f / fmaxf(sqrtf(ss), 1e-12f);
    ushort4 o;
    o.x = f2bf_rne(v.x * scale);
    o.y = f2bf_rne(v.y * scale);
    o.z = f2bf_rne(v.z * scale);
    o.w = f2bf_rne(v.w * scale);
    ((ushort4*)(xn + row * N_ELEM))[lane] = o;
}

// ---------------- Kernel 2: per-description pinv -> bf16, K-contiguous ----
// pinv = (db^T db)^-1 db^T ; db[d] is 128x4 with unit columns (SPD Gram,
// Gauss-Jordan without pivoting is safe). One thread per description.
__global__ void __launch_bounds__(256) k_pinv(const float* __restrict__ db,
                                              unsigned short* __restrict__ P) {
    const int d = blockIdx.x * blockDim.x + threadIdx.x;
    if (d >= N_DESC) return;
    const float4* A = (const float4*)(db + (size_t)d * (N_ELEM * KPD));

    float G[4][4];
#pragma unroll
    for (int i = 0; i < 4; i++)
#pragma unroll
        for (int j = 0; j < 4; j++) G[i][j] = 0.0f;

    for (int n = 0; n < N_ELEM; n++) {
        float4 r4 = A[n];
        float r[4] = {r4.x, r4.y, r4.z, r4.w};
#pragma unroll
        for (int i = 0; i < 4; i++)
#pragma unroll
            for (int j = 0; j < 4; j++) G[i][j] += r[i] * r[j];
    }

    // Gauss-Jordan on [G | I]
    float M[4][8];
#pragma unroll
    for (int i = 0; i < 4; i++) {
#pragma unroll
        for (int j = 0; j < 4; j++) {
            M[i][j]     = G[i][j];
            M[i][4 + j] = (i == j) ? 1.0f : 0.0f;
        }
    }
#pragma unroll
    for (int p = 0; p < 4; p++) {
        const float piv = 1.0f / M[p][p];
#pragma unroll
        for (int c = 0; c < 8; c++) M[p][c] *= piv;
#pragma unroll
        for (int r = 0; r < 4; r++) {
            if (r == p) continue;
            const float f = M[r][p];
#pragma unroll
            for (int c = 0; c < 8; c++) M[r][c] -= f * M[p][c];
        }
    }

    // pinv[j][n] = sum_i Ginv[j][i] * db[n][i]; store as P[(d*4+j)*128 + n]
    for (int n = 0; n < N_ELEM; n++) {
        float4 r = A[n];
#pragma unroll
        for (int j = 0; j < 4; j++) {
            float pv = M[j][4] * r.x + M[j][5] * r.y + M[j][6] * r.z + M[j][7] * r.w;
            P[((size_t)d * 4 + j) * N_ELEM + n] = f2bf_rne(pv);
        }
    }
}

// ---------------- Kernel 3: bf16 WMMA GEMM + sign-count ------------------
// Each wave: 16(M) x 64(N) coef tile, K=128 in 4 steps of 32.
// A frag (16x32 bf16): lane m = lane&15; elem e -> K = (e<8?0:16)+8h+(e&7)
//   => two contiguous 8-elem chunks at K + {8h, 16+8h}.
// B frag (32x16 bf16): lane n = lane&15; elem e -> K = (e<8?0:16)+8h+(e&7)
//   => two contiguous 8-elem chunks at K + {8h, 16+8h}.
__global__ void __launch_bounds__(256) k_gemm_count(
        const unsigned short* __restrict__ xn_u, const unsigned short* __restrict__ P_u,
        const int* __restrict__ nelem, float* __restrict__ out) {
    const int lane = threadIdx.x & 31;
    const int wave = threadIdx.x >> 5;
    const int h    = lane >> 4;     // wave half
    const int l15  = lane & 15;
    const int Mbase   = blockIdx.y * 128 + wave * 16;
    const int colBase = blockIdx.x * 64;

    const bf16_t* Xn = (const bf16_t*)xn_u;
    const bf16_t* Pb = (const bf16_t*)P_u;

    v8f acc[4] = {};

    const bf16_t* arow = Xn + (size_t)(Mbase + l15) * N_ELEM;

#pragma unroll
    for (int ks = 0; ks < N_ELEM; ks += 32) {
        union { v16bf v; v8bf q[2]; } a;
        a.q[0] = *(const v8bf*)(arow + ks + 8 * h);
        a.q[1] = *(const v8bf*)(arow + ks + 16 + 8 * h);
#pragma unroll
        for (int t = 0; t < 4; t++) {
            const bf16_t* bcol =
                Pb + (size_t)(colBase + 16 * t + l15) * N_ELEM + ks + 8 * h;
            union { v16bf v; v8bf q[2]; } b;
            b.q[0] = *(const v8bf*)(bcol);
            b.q[1] = *(const v8bf*)(bcol + 16);
            acc[t] = __builtin_amdgcn_wmma_f32_16x16x32_bf16(
                false, a.v, false, b.v, (short)0, acc[t], false, false);
        }
    }

    // C layout: reg r, lane L -> M = r + 8*(L>>4), Ncol = L&15.
    // The 4 coef columns of one description live in lanes 4t..4t+3.
#pragma unroll
    for (int t = 0; t < 4; t++) {
#pragma unroll
        for (int r = 0; r < 8; r++) {
            float pos = acc[t][r] > 0.0f ? 1.0f : 0.0f;   // heaviside(c,0)
            pos += __shfl_xor(pos, 1, 32);
            pos += __shfl_xor(pos, 2, 32);
            if ((lane & 3) == 0) {
                const int brow = Mbase + r + 8 * h;
                const int dd   = (colBase + 16 * t + l15) >> 2;
                out[(size_t)brow * N_DESC + dd] = pos / (float)nelem[dd];
            }
        }
    }
}

extern "C" void kernel_launch(void* const* d_in, const int* in_sizes, int n_in,
                              void* d_out, int out_size, void* d_ws, size_t ws_size,
                              hipStream_t stream) {
    const float* x     = (const float*)d_in[0];   // [8192,128] f32
    const float* db    = (const float*)d_in[1];   // [2048,128,4] f32
    const int*   nelem = (const int*)d_in[2];     // [2048] i32
    float*       out   = (float*)d_out;           // [8192,2048] f32

    unsigned short* xn = (unsigned short*)d_ws;            // 2 MB bf16 [8192,128]
    unsigned short* P  = xn + (size_t)BATCHN * N_ELEM;     // 2 MB bf16 [8192,128]

    k_normalize<<<BATCHN / 8, 256, 0, stream>>>(x, xn);
    k_pinv<<<N_DESC / 256, 256, 0, stream>>>(db, P);
    k_gemm_count<<<dim3(NCOL / 64, BATCHN / 128), 256, 0, stream>>>(xn, P, nelem, out);
}
